// Attention_45629732553202
// MI455X (gfx1250) — matrix operand order repaired and checked by reading, not compiled, not run on hardware
//
#include <hip/hip_runtime.h>
#include <cstdint>

constexpr int kB = 2, kT = 2048, kC = 2048;
constexpr int kH = 32, kHKV = 8, kD = 64, kGQ = kH / kHKV;
constexpr int kBT = kB * kT;

typedef __attribute__((ext_vector_type(16))) __bf16 v16bf;
typedef __attribute__((ext_vector_type(8)))  __bf16 v8bf;
typedef __attribute__((ext_vector_type(8)))  float  v8f;

__device__ inline v8f wmma_bf16(v16bf a, v16bf b, v8f c) {
  return __builtin_amdgcn_wmma_f32_16x16x32_bf16(false, a, false, b, (short)0, c,
                                                 false, false);
}

// A-matrix fragment (16x32 bf16): lane<16: K = ko+0..15; lane>=16: K = ko+16..31
__device__ inline v16bf load_a_frag(const __bf16* rowp, int ko, int lane) {
  const __bf16* p = rowp + ko + ((lane & 16) ? 16 : 0);
  union { v16bf v; v8bf h[2]; } u;
  u.h[0] = *(const v8bf*)(p);
  u.h[1] = *(const v8bf*)(p + 8);
  return u.v;
}

// B-matrix fragment (32x16 bf16): lane<16: K = ko+0..15; lane>=16: K = ko+16..31
__device__ inline v16bf load_b_frag(const __bf16* rowp, int ko, int lane) {
  const __bf16* p = rowp + ko + ((lane & 16) ? 16 : 0);
  union { v16bf v; v8bf h[2]; } u;
  u.h[0] = *(const v8bf*)(p);
  u.h[1] = *(const v8bf*)(p + 8);
  return u.v;
}

// ---- CDNA5 async global->LDS copy (ASYNCcnt-tracked, no VGPR round trip) ----
__device__ inline void async_ld128(__bf16* lds, const __bf16* g) {
  unsigned l = (unsigned)(uintptr_t)lds;  // low 32 bits of generic = LDS offset
  asm volatile("global_load_async_to_lds_b128 %0, %1, off"
               :: "v"(l), "v"(g) : "memory");
}
template <int N>
__device__ inline void wait_async() {
  asm volatile("s_wait_asynccnt %0" :: "n"(N) : "memory");
}

// ---------------- elementwise helpers ----------------

__global__ void cvt_f32_bf16(const float* __restrict__ s, __bf16* __restrict__ d,
                             size_t n) {
  size_t i = (size_t)blockIdx.x * blockDim.x + threadIdx.x;
  size_t st = (size_t)gridDim.x * blockDim.x;
  for (; i < n; i += st) d[i] = (__bf16)s[i];
}

// RoPE + relayout (B,T,nh,D) f32 -> (B,nh,T,D) bf16
__global__ void rope_kernel(const float* __restrict__ src, __bf16* __restrict__ dst,
                            int nh) {
  int idx = blockIdx.x * blockDim.x + threadIdx.x;
  int i = idx & 31;
  int t = (idx >> 5) % kT;
  int hh = ((idx >> 5) / kT) % nh;
  int b = (idx >> 5) / (kT * nh);
  if (b >= kB) return;
  float f = (float)t * __expf(-(float)i * (2.0f / 64.0f) * 9.210340371976184f);
  float c = __cosf(f), s = __sinf(f);
  const float* sp = src + ((size_t)(b * kT + t) * nh + hh) * kD;
  float x0 = sp[i], x1 = sp[i + 32];
  __bf16* dp = dst + ((size_t)(b * nh + hh) * kT + t) * kD;
  dp[i] = (__bf16)(x0 * c - x1 * s);
  dp[i + 32] = (__bf16)(x1 * c + x0 * s);
}

// (B,T,HKV,D) f32 -> (B,HKV,D,T) bf16
__global__ void vtrans_kernel(const float* __restrict__ v, __bf16* __restrict__ vt) {
  int idx = blockIdx.x * blockDim.x + threadIdx.x;
  int t = idx % kT;
  int d = (idx / kT) % kD;
  int h = (idx / (kT * kD)) % kHKV;
  int b = idx / (kT * kD * kHKV);
  if (b >= kB) return;
  vt[idx] = (__bf16)v[((size_t)(b * kT + t) * kHKV + h) * kD + d];
}

// -------- GEMM: C(f32, MxN) = A(bf16,MxK) * W(bf16,NxK)^T, LDS-staged --------
// Block tile 128M x 64N, K-step 64, double-buffered async-to-LDS pipeline.
constexpr int GM = 128, GN = 64, GK = 64;
constexpr int LDA = 72;  // padded LDS row stride: 144B = 36 banks, conflict-free

__global__ __launch_bounds__(256) void gemm_bf16_lds(
    const __bf16* __restrict__ A, const __bf16* __restrict__ W,
    float* __restrict__ Cm, int M, int N, int K) {
  __shared__ __bf16 lsA[2][GM * LDA];
  __shared__ __bf16 lsB[2][GN * LDA];
  int t = threadIdx.x;
  int lane = t & 31, wave = t >> 5;
  int ntn = N >> 6;
  int tm0 = (blockIdx.x / ntn) * GM;
  int tn0 = (blockIdx.x % ntn) * GN;

  auto issue_stage = [&](int ko, int buf) {
#pragma unroll
    for (int i = 0; i < 4; ++i) {             // A: 128x64 = 1024 chunks / 256 thr
      int c = t + i * 256;
      int row = c >> 3, col = (c & 7) * 8;
      async_ld128(&lsA[buf][row * LDA + col],
                  A + (size_t)(tm0 + row) * K + ko + col);
    }
#pragma unroll
    for (int i = 0; i < 2; ++i) {             // B: 64x64 = 512 chunks / 256 thr
      int c = t + i * 256;
      int row = c >> 3, col = (c & 7) * 8;
      async_ld128(&lsB[buf][row * LDA + col],
                  W + (size_t)(tn0 + row) * K + ko + col);
    }
  };

  v8f acc[4] = {};
  int nk = K / GK;
  issue_stage(0, 0);
  for (int kk = 0; kk < nk; ++kk) {
    // Unconditional prefetch of the next stage (clamped on the last iteration;
    // it targets buf[(kk+1)&1], which is never read again -> harmless).
    int kon = (kk + 1 < nk) ? (kk + 1) * GK : (nk - 1) * GK;
    issue_stage(kon, (kk + 1) & 1);
    wait_async<6>();   // 6 newer copies in flight; current stage complete
    __syncthreads();

    const __bf16* la = &lsA[kk & 1][(wave * 16 + (lane & 15)) * LDA];
    // Batch all fragment loads of the stage, then run 8 back-to-back WMMAs so
    // the ds_load clause drains under the matrix pipe.
    v16bf af0 = load_a_frag(la, 0, lane);
    v16bf af1 = load_a_frag(la, 32, lane);
    v16bf bfr[8];
#pragma unroll
    for (int s = 0; s < 4; ++s) {
      const __bf16* lb = &lsB[kk & 1][(s * 16 + (lane & 15)) * LDA];
      bfr[s]     = load_b_frag(lb, 0, lane);
      bfr[4 + s] = load_b_frag(lb, 32, lane);
    }
#pragma unroll
    for (int s = 0; s < 4; ++s) acc[s] = wmma_bf16(af0, bfr[s], acc[s]);
#pragma unroll
    for (int s = 0; s < 4; ++s) acc[s] = wmma_bf16(af1, bfr[4 + s], acc[s]);

    __syncthreads();  // all waves done reading buf (kk&1) before it refills
  }

  int mb = tm0 + wave * 16 + ((lane & 16) ? 8 : 0);
  int nb = tn0 + (lane & 15);
#pragma unroll
  for (int s = 0; s < 4; ++s)
#pragma unroll
    for (int r = 0; r < 8; ++r)
      Cm[(size_t)(mb + r) * N + s * 16 + nb] = acc[s][r];
}

// ---------------- Flash attention (transposed orientation, pipelined) --------
// Wave owns 16 q rows. Per 32-k chunk: S^T = K_rows*Q^T (4 wmma), online softmax
// (in-lane + one shfl_xor(16)), O^T += V^T*P^T (4 wmma). K-fragments for the
// next chunk are prefetched into registers; V^T fragments issued at loop top.

__global__ __launch_bounds__(256) void attn_kernel(
    const __bf16* __restrict__ qb, const __bf16* __restrict__ kb,
    const __bf16* __restrict__ vt, __bf16* __restrict__ yb) {
  int lane = threadIdx.x & 31;
  int wave = threadIdx.x >> 5;
  int bid = blockIdx.x;
  int t128 = bid % (kT / 128);
  int h = (bid / (kT / 128)) % kH;
  int b = bid / ((kT / 128) * kH);
  int hkv = h / kGQ;
  int q0 = t128 * 128 + wave * 16;
  int qrow = q0 + (lane & 15);
  const __bf16* qp = qb + ((size_t)(b * kH + h) * kT) * kD;
  const __bf16* kp = kb + ((size_t)(b * kHKV + hkv) * kT) * kD;
  const __bf16* vp = vt + ((size_t)(b * kHKV + hkv) * kD) * kT;

  v16bf qB0 = load_b_frag(qp + (size_t)qrow * kD, 0, lane);
  v16bf qB1 = load_b_frag(qp + (size_t)qrow * kD, 32, lane);

  float m = -1e30f, l = 0.0f;
  v8f acc[4] = {};
  bool lo_half = (lane & 16) == 0;
  int nch = (q0 + 16 + 31) >> 5;

  // preload K fragments for chunk 0
  const __bf16* kr0 = kp + (size_t)(lane & 15) * kD;
  const __bf16* kr1 = kp + (size_t)(16 + (lane & 15)) * kD;
  v16bf ka00 = load_a_frag(kr0, 0, lane), ka01 = load_a_frag(kr0, 32, lane);
  v16bf ka10 = load_a_frag(kr1, 0, lane), ka11 = load_a_frag(kr1, 32, lane);

  for (int c = 0; c < nch; ++c) {
    int j = c * 32;
    // V^T fragments for this chunk: issue early, consumed after softmax VALU
    v16bf vf0 = load_a_frag(vp + (size_t)(0 * 16 + (lane & 15)) * kT, j, lane);
    v16bf vf1 = load_a_frag(vp + (size_t)(1 * 16 + (lane & 15)) * kT, j, lane);
    v16bf vf2 = load_a_frag(vp + (size_t)(2 * 16 + (lane & 15)) * kT, j, lane);
    v16bf vf3 = load_a_frag(vp + (size_t)(3 * 16 + (lane & 15)) * kT, j, lane);

    v8f s0 = {}, s1 = {};
    s0 = wmma_bf16(ka00, qB0, s0);
    s0 = wmma_bf16(ka01, qB1, s0);
    s1 = wmma_bf16(ka10, qB0, s1);
    s1 = wmma_bf16(ka11, qB1, s1);

    // prefetch next chunk's K fragments (clamped; masked out anyway)
    int j2 = j + 32;
    if (j2 > kT - 32) j2 = kT - 32;
    const __bf16* nr0 = kp + (size_t)(j2 + (lane & 15)) * kD;
    const __bf16* nr1 = kp + (size_t)(j2 + 16 + (lane & 15)) * kD;
    ka00 = load_a_frag(nr0, 0, lane);
    ka01 = load_a_frag(nr0, 32, lane);
    ka10 = load_a_frag(nr1, 0, lane);
    ka11 = load_a_frag(nr1, 32, lane);

    int krb = j + ((lane & 16) ? 8 : 0);
    float p0[8], p1[8], cmax = -1e30f;
#pragma unroll
    for (int r = 0; r < 8; ++r) {
      float a0 = s0[r] * 0.125f + ((krb + r) > qrow ? -1e9f : 0.0f);
      float a1 = s1[r] * 0.125f + ((krb + 16 + r) > qrow ? -1e9f : 0.0f);
      p0[r] = a0; p1[r] = a1;
      cmax = fmaxf(cmax, fmaxf(a0, a1));
    }
    cmax = fmaxf(cmax, __shfl_xor(cmax, 16, 32));
    float mn = fmaxf(m, cmax);
    float alpha = __expf(m - mn);
    float sum = 0.0f;
    float e0[8], e1[8];
#pragma unroll
    for (int r = 0; r < 8; ++r) {
      e0[r] = __expf(p0[r] - mn);
      e1[r] = __expf(p1[r] - mn);
      sum += e0[r] + e1[r];
    }
    sum += __shfl_xor(sum, 16, 32);
    l = l * alpha + sum;
    m = mn;

    union { v16bf v; v8bf hlf[2]; } pf;
#pragma unroll
    for (int r = 0; r < 8; ++r) {
      float o0 = __shfl_xor(e0[r], 16, 32);
      float o1 = __shfl_xor(e1[r], 16, 32);
      pf.v[r]     = (__bf16)(lo_half ? e0[r] : o1);
      pf.v[8 + r] = (__bf16)(lo_half ? o0 : e1[r]);
    }
#pragma unroll
    for (int dt = 0; dt < 4; ++dt)
#pragma unroll
      for (int r = 0; r < 8; ++r) acc[dt][r] *= alpha;
    acc[0] = wmma_bf16(vf0, pf.v, acc[0]);
    acc[1] = wmma_bf16(vf1, pf.v, acc[1]);
    acc[2] = wmma_bf16(vf2, pf.v, acc[2]);
    acc[3] = wmma_bf16(vf3, pf.v, acc[3]);
  }

  float inv = 1.0f / l;
  __bf16* yp = yb + ((size_t)(b * kT + qrow) * kH + h) * kD;
#pragma unroll
  for (int dt = 0; dt < 4; ++dt) {
    int db = dt * 16 + ((lane & 16) ? 8 : 0);
#pragma unroll
    for (int r = 0; r < 8; ++r)
      yp[db + r] = (__bf16)(acc[dt][r] * inv);
  }
}

// ---------------- host launch ----------------

extern "C" void kernel_launch(void* const* d_in, const int* in_sizes, int n_in,
                              void* d_out, int out_size, void* d_ws, size_t ws_size,
                              hipStream_t stream) {
  (void)in_sizes; (void)n_in; (void)out_size; (void)ws_size;
  const float* x  = (const float*)d_in[0];
  const float* wq = (const float*)d_in[2];
  const float* wk = (const float*)d_in[3];
  const float* wv = (const float*)d_in[4];
  const float* wc = (const float*)d_in[5];
  float* out = (float*)d_out;

  char* ws = (char*)d_ws;
  size_t off = 0;
  auto take = [&](size_t bytes) -> void* {
    void* p = ws + off;
    off = (off + bytes + 255) & ~(size_t)255;
    return p;
  };
  __bf16* xb  = (__bf16*)take((size_t)kBT * kC * 2);
  __bf16* wqb = (__bf16*)take((size_t)kC * kC * 2);
  __bf16* wkb = (__bf16*)take((size_t)kHKV * kD * kC * 2);
  __bf16* wvb = (__bf16*)take((size_t)kHKV * kD * kC * 2);
  __bf16* wcb = (__bf16*)take((size_t)kC * kC * 2);
  float*  qf  = (float*)take((size_t)kBT * kC * 4);
  float*  kf  = (float*)take((size_t)kBT * kHKV * kD * 4);
  float*  vf  = (float*)take((size_t)kBT * kHKV * kD * 4);
  __bf16* qbh = (__bf16*)take((size_t)kB * kH * kT * kD * 2);
  __bf16* kbh = (__bf16*)take((size_t)kB * kHKV * kT * kD * 2);
  __bf16* vtb = (__bf16*)take((size_t)kB * kHKV * kD * kT * 2);
  __bf16* ybf = (__bf16*)take((size_t)kBT * kC * 2);

  cvt_f32_bf16<<<2048, 256, 0, stream>>>(x, xb, (size_t)kBT * kC);
  cvt_f32_bf16<<<2048, 256, 0, stream>>>(wq, wqb, (size_t)kC * kC);
  cvt_f32_bf16<<<1024, 256, 0, stream>>>(wk, wkb, (size_t)kHKV * kD * kC);
  cvt_f32_bf16<<<1024, 256, 0, stream>>>(wv, wvb, (size_t)kHKV * kD * kC);
  cvt_f32_bf16<<<2048, 256, 0, stream>>>(wc, wcb, (size_t)kC * kC);

  gemm_bf16_lds<<<(kBT / GM) * (kC / GN), 256, 0, stream>>>(
      xb, wqb, qf, kBT, kC, kC);
  gemm_bf16_lds<<<(kBT / GM) * ((kHKV * kD) / GN), 256, 0, stream>>>(
      xb, wkb, kf, kBT, kHKV * kD, kC);
  gemm_bf16_lds<<<(kBT / GM) * ((kHKV * kD) / GN), 256, 0, stream>>>(
      xb, wvb, vf, kBT, kHKV * kD, kC);

  rope_kernel<<<(kB * kT * kH * 32) / 256, 256, 0, stream>>>(qf, qbh, kH);
  rope_kernel<<<(kB * kT * kHKV * 32) / 256, 256, 0, stream>>>(kf, kbh, kHKV);
  vtrans_kernel<<<(kB * kHKV * kD * kT) / 256, 256, 0, stream>>>(vf, vtb);

  attn_kernel<<<kB * kH * (kT / 128), 256, 0, stream>>>(qbh, kbh, vtb, ybf);

  gemm_bf16_lds<<<(kBT / GM) * (kC / GN), 256, 0, stream>>>(
      ybf, wcb, out, kBT, kC, kC);
}
